// Attention_20822001451163
// MI455X (gfx1250) — matrix operand order repaired and checked by reading, not compiled, not run on hardware
//
#include <hip/hip_runtime.h>
#include <hip/hip_bf16.h>

typedef unsigned short u16;
typedef __attribute__((ext_vector_type(16))) __bf16 v16bf;
typedef __attribute__((ext_vector_type(8)))  float  v8f;

#define NEG_INF_F (-__builtin_huge_valf())

// ---------- bf16 helpers (round-to-nearest-even) ----------
__device__ __forceinline__ u16 f2bf(float f) {
  unsigned int u = __float_as_uint(f);
  if ((u & 0x7FFFFFFFu) > 0x7F800000u) return (u16)((u >> 16) | 0x40u); // NaN
  unsigned int r = u + 0x7FFFu + ((u >> 16) & 1u);
  return (u16)(r >> 16);
}
__device__ __forceinline__ float bf2f(u16 h) {
  return __uint_as_float(((unsigned int)h) << 16);
}

// ---------- CDNA5 async memory->LDS (ASYNCcnt-tracked DMA, no VGPR round trip) ----------
__device__ __forceinline__ void async_load_b128(unsigned lds_byte_off, unsigned long long gaddr) {
  asm volatile("global_load_async_to_lds_b128 %0, %1, off"
               :: "v"(lds_byte_off), "v"(gaddr)
               : "memory");
}
__device__ __forceinline__ void wait_asynccnt0() {
  asm volatile("s_wait_asynccnt 0x0" ::: "memory");
}
// LDS byte offset = low 32 bits of flat address
__device__ __forceinline__ unsigned lds_off(const void* p) {
  return (unsigned)(size_t)p;
}

// ---------- Kernel 0: f32 -> bf16 conversion (vectorized) ----------
__global__ __launch_bounds__(256)
void f32_to_bf16_kernel(const float* __restrict__ src, u16* __restrict__ dst, int n4) {
  int i = blockIdx.x * blockDim.x + threadIdx.x;
  int stride = gridDim.x * blockDim.x;
  for (; i < n4; i += stride) {
    float4 f = ((const float4*)src)[i];
    unsigned int lo = (unsigned int)f2bf(f.x) | ((unsigned int)f2bf(f.y) << 16);
    unsigned int hi = (unsigned int)f2bf(f.z) | ((unsigned int)f2bf(f.w) << 16);
    ((uint2*)dst)[i] = make_uint2(lo, hi);
  }
}

// ---------- Templated WMMA GEMM (double-buffered, async-DMA staged) ----------
// C[m][n] = sum_k A[m][k] * Bop[k][n]
//   BKN == 0 : B row-major [N][K] (weights / key matrix) -> contiguous along k, async DMA
//   BKN == 1 : B row-major [K][N] (V matrix)             -> packed-pair transpose through VGPRs
enum { EPI_BF16 = 0, EPI_MASK_BF16 = 1, EPI_BIAS_F32 = 2 };

template <int BKN, int EPI>
__global__ __launch_bounds__(256)
void gemm_wmma(const u16* __restrict__ A, const u16* __restrict__ B,
               void* __restrict__ Cv,
               int K, int lda, int ldb, int ldc,
               long long Az, long long Bz, long long Cz,
               const float* __restrict__ aux, long long auxZ)
{
  // 128x128 block tile, BK=32, 8 waves in 4x2; wave tile 32x64 (2x4 wmma frags)
  __shared__ u16 As[2][128 * 40];   // [buf][row*40 + k], 80B row stride (16B aligned, bank-skewed)
  __shared__ u16 Bs[2][128 * 40];   // [buf][n*40 + k]
  const unsigned BUFB = 128 * 40 * 2;  // buffer stride in bytes

  const int tid  = threadIdx.x;
  const int lane = tid & 31;
  const int wave = tid >> 5;
  const int wrow = wave >> 1;    // 0..3
  const int wcol = wave & 1;     // 0..1
  const int m0 = blockIdx.x * 128;
  const int n0 = blockIdx.y * 128;
  const int z  = blockIdx.z;
  const int lane_ml = lane & 15;
  const int lane_hi = lane >> 4;

  A += (long long)z * Az;
  B += (long long)z * Bz;
  const float* auxp = aux ? (aux + (long long)z * auxZ) : nullptr;

  const v8f vzero = {0.f,0.f,0.f,0.f,0.f,0.f,0.f,0.f};
  v8f acc[2][4];
#pragma unroll
  for (int i = 0; i < 2; ++i)
#pragma unroll
    for (int j = 0; j < 4; ++j) acc[i][j] = vzero;

  union Frag { struct { uint4 lo, hi; } u; v16bf v; };
  struct BV { uint4 r0, r1; };   // two consecutive k-rows for transpose staging

  // --- persistent staging state: live u64 global addresses, bumped per K-step ---
  const int r0 = tid >> 2;             // 0..63
  const int cg = (tid & 3) << 3;       // 0,8,16,24
  unsigned long long aAddr0 = (unsigned long long)(size_t)(A + (long long)(m0 + r0)      * lda + cg);
  unsigned long long aAddr1 = (unsigned long long)(size_t)(A + (long long)(m0 + r0 + 64) * lda + cg);
  const unsigned ldsA0 = lds_off(&As[0][r0 * 40 + cg]);
  const unsigned ldsA1 = lds_off(&As[0][(r0 + 64) * 40 + cg]);

  unsigned long long bAddr0 = 0, bAddr1 = 0;
  unsigned ldsB0 = 0, ldsB1 = 0;
  const int t_kk = (tid >> 4) << 1;    // 0,2,..,30   (BKN==1)
  const int t_ng = (tid & 15) << 3;    // 0..120      (BKN==1)
  if (BKN == 0) {
    bAddr0 = (unsigned long long)(size_t)(B + (long long)(n0 + r0)      * ldb + cg);
    bAddr1 = (unsigned long long)(size_t)(B + (long long)(n0 + r0 + 64) * ldb + cg);
    ldsB0 = lds_off(&Bs[0][r0 * 40 + cg]);
    ldsB1 = lds_off(&Bs[0][(r0 + 64) * 40 + cg]);
  } else {
    bAddr0 = (unsigned long long)(size_t)(B + (long long)t_kk       * ldb + (n0 + t_ng));
    bAddr1 = (unsigned long long)(size_t)(B + (long long)(t_kk + 1) * ldb + (n0 + t_ng));
  }
  const unsigned long long bStepT = (unsigned long long)ldb * 64ull; // 32 rows * 2B (BKN==1)

  auto stageA = [&](int buf) {
    unsigned boff = (unsigned)buf * BUFB;
    async_load_b128(ldsA0 + boff, aAddr0); aAddr0 += 64;  // BK=32 bf16 = 64B
    async_load_b128(ldsA1 + boff, aAddr1); aAddr1 += 64;
  };
  auto stageB_async = [&](int buf) {     // BKN == 0
    unsigned boff = (unsigned)buf * BUFB;
    async_load_b128(ldsB0 + boff, bAddr0); bAddr0 += 64;
    async_load_b128(ldsB1 + boff, bAddr1); bAddr1 += 64;
  };
  auto loadB_T = [&]() -> BV {           // BKN == 1
    BV bv;
    bv.r0 = *(const uint4*)(size_t)bAddr0; bAddr0 += bStepT;
    bv.r1 = *(const uint4*)(size_t)bAddr1; bAddr1 += bStepT;
    return bv;
  };
  auto storeB_T = [&](int buf, BV bv) {
    union { uint4 u; u16 s[8]; } c0, c1;
    c0.u = bv.r0; c1.u = bv.r1;
#pragma unroll
    for (int j = 0; j < 8; ++j) {
      unsigned pk = (unsigned)c0.s[j] | ((unsigned)c1.s[j] << 16); // (k, k+1) pair
      *(unsigned*)(&Bs[buf][(t_ng + j) * 40 + t_kk]) = pk;         // 4B-aligned (t_kk even)
    }
  };

  auto compute = [&](int buf) {
    // Preload ALL fragments into distinct registers first so the 8 WMMAs can
    // issue back-to-back without ds_load register-reuse stalls.
    Frag a[2], b[4];
#pragma unroll
    for (int sm = 0; sm < 2; ++sm) {
      int row = wrow * 32 + sm * 16 + lane_ml;
      int kb  = lane_hi * 8;
      a[sm].u.lo = *(const uint4*)(&As[buf][row * 40 + kb]);
      a[sm].u.hi = *(const uint4*)(&As[buf][row * 40 + kb + 16]);
    }
#pragma unroll
    for (int j = 0; j < 4; ++j) {
      int nn  = wcol * 64 + j * 16 + lane_ml;
      int kb2 = lane_hi * 8;
      b[j].u.lo = *(const uint4*)(&Bs[buf][nn * 40 + kb2]);
      b[j].u.hi = *(const uint4*)(&Bs[buf][nn * 40 + kb2 + 16]);
    }
#pragma unroll
    for (int j = 0; j < 4; ++j)
#pragma unroll
      for (int sm = 0; sm < 2; ++sm)
        acc[sm][j] = __builtin_amdgcn_wmma_f32_16x16x32_bf16(
            false, a[sm].v, false, b[j].v, (short)0, acc[sm][j], false, false);
  };

  // --- prologue: stage tile 0 into buffer 0 ---
  stageA(0);
  if (BKN == 0) {
    stageB_async(0);
  } else {
    BV bv = loadB_T();
    storeB_T(0, bv);
  }

  const int nk = K >> 5;
  for (int kt = 0; kt < nk; ++kt) {
    const int cur = kt & 1;
    const int nxt = cur ^ 1;
    wait_asynccnt0();        // tile kt's async DMA complete (this wave)
    __syncthreads();         // all waves' DMAs + ds stores visible; prev-prev buffer free

    BV bvn;
    const bool has_next = (kt + 1 < nk);
    if (has_next) {
      stageA(nxt);                              // overlaps with WMMA below
      if (BKN == 0) stageB_async(nxt);
      else          bvn = loadB_T();
    }

    compute(cur);

    if (BKN == 1 && has_next) storeB_T(nxt, bvn); // dscnt waited by next __syncthreads
  }

  // ---- epilogue (C layout: VGPR i -> M = i + (lane>=16 ? 8 : 0), N = lane&15)
#pragma unroll
  for (int sm = 0; sm < 2; ++sm) {
#pragma unroll
    for (int j = 0; j < 4; ++j) {
#pragma unroll
      for (int i = 0; i < 8; ++i) {
        int grow = m0 + wrow * 32 + sm * 16 + lane_hi * 8 + i;
        int gcol = n0 + wcol * 64 + j * 16 + lane_ml;
        float v = acc[sm][j][i];
        long long idx = (long long)z * Cz + (long long)grow * ldc + gcol;
        if (EPI == EPI_BF16) {
          ((u16*)Cv)[idx] = f2bf(v);
        } else if (EPI == EPI_MASK_BF16) {
          float val = v * 0.03125f;                 // 1/sqrt(1024)
          if (gcol > grow) val = NEG_INF_F;         // causal
          if (auxp[grow] * auxp[gcol] != 1.0f) val = NEG_INF_F;  // padding
          ((u16*)Cv)[idx] = f2bf(val);
        } else {
          ((float*)Cv)[idx] = v + auxp[gcol];
        }
      }
    }
  }
}

// ---------- Softmax over rows of bf16 logits (in place, per-element) ----------
__global__ __launch_bounds__(256)
void softmax_rows(u16* __restrict__ logits, int S) {
  long long row = blockIdx.x;
  u16* lr = logits + row * (long long)S;
  int t = threadIdx.x;

  float vals[8];
  float m = NEG_INF_F;
#pragma unroll
  for (int i = 0; i < 8; ++i) {
    vals[i] = bf2f(lr[t + (i << 8)]);
    m = fmaxf(m, vals[i]);
  }
#pragma unroll
  for (int off = 16; off > 0; off >>= 1) m = fmaxf(m, __shfl_xor(m, off, 32));

  __shared__ float redmax[8];
  __shared__ float redsum[8];
  if ((t & 31) == 0) redmax[t >> 5] = m;
  __syncthreads();
  float M = redmax[0];
#pragma unroll
  for (int w = 1; w < 8; ++w) M = fmaxf(M, redmax[w]);

  float s = 0.f;
#pragma unroll
  for (int i = 0; i < 8; ++i) { vals[i] = __expf(vals[i] - M); s += vals[i]; }
#pragma unroll
  for (int off = 16; off > 0; off >>= 1) s += __shfl_xor(s, off, 32);
  if ((t & 31) == 0) redsum[t >> 5] = s;
  __syncthreads();
  float T = 0.f;
#pragma unroll
  for (int w = 0; w < 8; ++w) T += redsum[w];
  float inv = 1.0f / T;
#pragma unroll
  for (int i = 0; i < 8; ++i) lr[t + (i << 8)] = f2bf(vals[i] * inv);
}

// ---------- host ----------
extern "C" void kernel_launch(void* const* d_in, const int* in_sizes, int n_in,
                              void* d_out, int out_size, void* d_ws, size_t ws_size,
                              hipStream_t stream) {
  (void)in_sizes; (void)n_in; (void)out_size; (void)ws_size;
  const float* x    = (const float*)d_in[0];
  const float* mask = (const float*)d_in[1];
  const float* Wq   = (const float*)d_in[2];
  const float* Wk   = (const float*)d_in[3];
  const float* Wv   = (const float*)d_in[4];
  const float* Wo   = (const float*)d_in[5];
  const float* bo   = (const float*)d_in[6];
  float* out = (float*)d_out;

  const int Bn = 4, S = 2048, D = 1024;
  const long long MS = (long long)Bn * S;        // 8192 tokens

  // workspace layout (u16 units): ~120 MB total
  u16* ws16 = (u16*)d_ws;
  u16* xb  = ws16;                 // [MS][D]
  u16* wqb = xb  + MS * D;         // [D][D]
  u16* wkb = wqb + (long long)D * D;
  u16* wvb = wkb + (long long)D * D;
  u16* wob = wvb + (long long)D * D;
  u16* qb  = wob + (long long)D * D;   // [MS][D]
  u16* kb  = qb  + MS * D;
  u16* vb  = kb  + MS * D;
  u16* ab  = vb  + MS * D;             // attn_out [MS][D]
  u16* pb  = ab  + MS * D;             // logits/probs [B][S][S]

  dim3 blk(256);

  // 0) convert to bf16
  f32_to_bf16_kernel<<<512, blk, 0, stream>>>(x,  xb,  (int)(MS * D / 4));
  f32_to_bf16_kernel<<<256, blk, 0, stream>>>(Wq, wqb, D * D / 4);
  f32_to_bf16_kernel<<<256, blk, 0, stream>>>(Wk, wkb, D * D / 4);
  f32_to_bf16_kernel<<<256, blk, 0, stream>>>(Wv, wvb, D * D / 4);
  f32_to_bf16_kernel<<<256, blk, 0, stream>>>(Wo, wob, D * D / 4);

  // 1) q/k/v projections: y = x @ W^T  (W row-major [out][in] == [N][K])
  dim3 g1((unsigned)(MS / 128), D / 128, 1);
  gemm_wmma<0, EPI_BF16><<<g1, blk, 0, stream>>>(xb, wqb, qb, D, D, D, D, 0, 0, 0, nullptr, 0);
  gemm_wmma<0, EPI_BF16><<<g1, blk, 0, stream>>>(xb, wkb, kb, D, D, D, D, 0, 0, 0, nullptr, 0);
  gemm_wmma<0, EPI_BF16><<<g1, blk, 0, stream>>>(xb, wvb, vb, D, D, D, D, 0, 0, 0, nullptr, 0);

  // 2) logits = q @ k^T per batch; fused scale + causal + padding mask
  dim3 g2(S / 128, S / 128, Bn);
  gemm_wmma<0, EPI_MASK_BF16><<<g2, blk, 0, stream>>>(
      qb, kb, pb, D, D, D, S,
      (long long)S * D, (long long)S * D, (long long)S * S, mask, S);

  // 3) softmax rows (in place)
  softmax_rows<<<(unsigned)MS, blk, 0, stream>>>(pb, S);

  // 4) attn_out = P @ V per batch (V is [K][N] -> transpose-staged B)
  dim3 g4(S / 128, D / 128, Bn);
  gemm_wmma<1, EPI_BF16><<<g4, blk, 0, stream>>>(
      pb, vb, ab, S, S, D, D,
      (long long)S * S, (long long)S * D, (long long)S * D, nullptr, 0);

  // 5) out = attn_out @ Wo^T + bo  (f32 output)
  dim3 g5((unsigned)(MS / 128), D / 128, 1);
  gemm_wmma<0, EPI_BIAS_F32><<<g5, blk, 0, stream>>>(
      ab, wob, out, D, D, D, D, 0, 0, 0, bo, 0);
}
